// Atoms_64725157151151
// MI455X (gfx1250) — matrix operand order repaired and checked by reading, not compiled, not run on hardware
//
#include <hip/hip_runtime.h>
#include <hip/hip_bf16.h>
#include <math.h>
#include <stdint.h>

// ---------------------------------------------------------------------------
// DDSP-style synth forward for MI455X (gfx1250, wave32).
//   B=32, C=128, frames=128, samples=32768, 16 harmonics (17 channels).
// WMMA f16 (v_wmma_f32_16x16x32_f16) for: MLP stacks, conv_upsample conv3
// stages (3 shifted GEMMs, 2x-upsample folded into the B gather), and the
// 64-tap FIR (fft_convolve == causal FIR since filt has 64 nonzero taps).
// osc kept as f16 in workspace (35.7 MB) -> dominant HBM traffic ~4us @23.3TB/s.
// bl_noise never materialized (FIR fused with the final combine).
// FIR taps staged LDS-side via global_load_async_to_lds_b128 / s_wait_asynccnt.
// ---------------------------------------------------------------------------

typedef _Float16 v16h __attribute__((ext_vector_type(16)));
typedef float    v8f  __attribute__((ext_vector_type(8)));

#define B_SZ     32
#define C_SZ     128
#define NFRAMES  128
#define NSAMP    32768
#define NCH      17            // 1 + 16 harmonics
#define NNOTES   128
#define PI_F     3.14159265358979f

// ---- input index map (setup_inputs insertion-order flatten) ----------------
#define IN_X         0
#define IN_F0_W(i)   (1 + (i))      // 4x (128,128)
#define IN_F0_B(i)   (5 + (i))      // 4x (128,)
#define IN_SS_W(i)   (9 + (i))      // 3x (128,128) + (128,16)
#define IN_SS_B(i)   (13 + (i))     // 3x (128,) + (16,)
#define IN_CU_BASE(h) (17 + 12*(h)) // h: 0=f0_change 1=amp 2=decay 3=mix
#define IN_NOISE_LVL 65
#define IN_CFREQ     66
#define IN_HARM      67
#define IN_NOISE     68
#define IN_GUMBEL    69

// ---- workspace layout (bytes) ----------------------------------------------
#define WS_OSC      0u                                  // f16 B*17*NSAMP
#define WS_OSC_SZ   (B_SZ * NCH * NSAMP * 2u)
#define WS_F0VAL    (WS_OSC + WS_OSC_SZ)                // f32 B
#define WS_SPEC     (WS_F0VAL + 128u)                   // f32 B*16
#define WS_F0CF     (WS_SPEC + 2048u)                   // f32 B*128
#define WS_AMPF     (WS_F0CF + 16384u)
#define WS_DECF     (WS_AMPF + 16384u)
#define WS_MIXF     (WS_DECF + 16384u)

// ---- WMMA fragment index helpers (CDNA5 ISA §7.12.2, wave32) ----------
// A (16x32, f16): lane m=lane&15, half=lane>>4; elem i -> k offset:
__device__ __forceinline__ int a16_k(int i, int half) {
  int v = i >> 1, sub = i & 1;
  return 2 * (v & 3) + 16 * (v >> 2) + 8 * half + sub;
}
// B (32x16, f16): lane n=lane&15, half=lane>>4; elem i -> k offset:
__device__ __forceinline__ int b16_k(int i, int half) {
  int v = i >> 1, sub = i & 1;
  return 2 * (v & 3) + 16 * (v >> 2) + 8 * half + sub;
}
// C/D (16x16, f32): elem r -> row m = r + 8*half, col n = lane&15.

__device__ __forceinline__ float leaky(float x) { return x >= 0.f ? x : 0.2f * x; }
__device__ __forceinline__ float clamp01(float x) { return fminf(fmaxf(x, 0.f), 1.f); }

// ===========================================================================
// Kernel 1: both MLP stacks (f0 + spec_shape) via WMMA; 1 block, 256 thr.
// ===========================================================================
struct MlpArgs {
  const float* x;
  const float* fw[4]; const float* fb[4];
  const float* sw[4]; const float* sb[4];
  const float* gumbel; const float* cfreq;
  float* f0_val;   // (B)
  float* spec;     // (B,16) clamped
};

// HIDDEN=true: leaky + f16 store; HIDDEN=false: raw f32 store (no branches).
template <bool HIDDEN>
__device__ void dense_wmma(const _Float16* __restrict__ in,
                           const float* __restrict__ W,
                           const float* __restrict__ bias, int dout,
                           _Float16* __restrict__ out16,
                           float* __restrict__ out32) {
  const int lane = threadIdx.x & 31, wv = threadIdx.x >> 5;
  const int half = lane >> 4, mr = lane & 15, nn = lane & 15;
  const int ntiles = dout >> 4;
  for (int tile = wv; tile < 2 * ntiles; tile += 8) {
    const int mt = tile / ntiles, nt = tile % ntiles;
    v8f acc = {};
    for (int ks = 0; ks < 4; ++ks) {
      v16h a, b;
      const int k0 = ks * 32;
#pragma unroll
      for (int i = 0; i < 16; ++i) {
        a[i] = in[(mt * 16 + mr) * C_SZ + k0 + a16_k(i, half)];
        b[i] = (_Float16)W[(k0 + b16_k(i, half)) * dout + nt * 16 + nn];
      }
      acc = __builtin_amdgcn_wmma_f32_16x16x32_f16(false, a, false, b,
                                                   (short)0, acc, false, false);
    }
#pragma unroll
    for (int r = 0; r < 8; ++r) {
      const int m = mt * 16 + r + 8 * half;
      const int n = nt * 16 + nn;
      const float v = acc[r] + bias[n];
      if (HIDDEN) out16[m * dout + n] = (_Float16)leaky(v);
      else        out32[m * dout + n] = v;
    }
  }
}

__global__ void mlp_kernel(MlpArgs A) {
  __shared__ _Float16 actX[B_SZ * C_SZ], actA[B_SZ * C_SZ], actB[B_SZ * C_SZ];
  __shared__ float zbuf[B_SZ * C_SZ];
  const int tid = threadIdx.x;
  for (int i = tid; i < B_SZ * C_SZ; i += 256) actX[i] = (_Float16)A.x[i];
  __syncthreads();

  // f0 stack: 3x leaky hidden + linear out
  dense_wmma<true >(actX, A.fw[0], A.fb[0], 128, actA, nullptr); __syncthreads();
  dense_wmma<true >(actA, A.fw[1], A.fb[1], 128, actB, nullptr); __syncthreads();
  dense_wmma<true >(actB, A.fw[2], A.fb[2], 128, actA, nullptr); __syncthreads();
  dense_wmma<false>(actA, A.fw[3], A.fb[3], 128, nullptr, zbuf); __syncthreads();

  // gumbel-hard argmax -> f0 = center_freqs[argmax] (ST trick value == hard)
  if (tid < B_SZ) {
    float best = -1e30f; int bi = 0;
    for (int j = 0; j < NNOTES; ++j) {
      const float v = zbuf[tid * NNOTES + j] + A.gumbel[tid * NNOTES + j];
      if (v > best) { best = v; bi = j; }
    }
    A.f0_val[tid] = A.cfreq[bi];
  }
  __syncthreads();

  // spec_shape stack (dout 16 at the end), unit_act == clamp in fwd
  dense_wmma<true >(actX, A.sw[0], A.sb[0], 128, actA, nullptr); __syncthreads();
  dense_wmma<true >(actA, A.sw[1], A.sb[1], 128, actB, nullptr); __syncthreads();
  dense_wmma<true >(actB, A.sw[2], A.sb[2], 128, actA, nullptr); __syncthreads();
  dense_wmma<false>(actA, A.sw[3], A.sb[3], 16, nullptr, zbuf);  __syncthreads();
  if (tid < B_SZ * 16) A.spec[tid] = clamp01(zbuf[tid]);
}

// ===========================================================================
// Kernel 2: conv_upsample head; grid = B, 256 thr. conv3 = 3 shifted GEMMs
// (WMMA); 2x nearest upsample folded into the B-fragment gather (q>>1).
// A fragment hoisted out of the n-tile loop (1 weight load -> ntiles WMMAs).
// ===========================================================================
struct ConvArgs {
  const float* x;                      // (B,128)
  const float* lin_w; const float* lin_b;
  const float* cw[4]; const float* cb[4];
  const float* ow; const float* ob;
  float* outF;                         // (B,128) activated frames
  int act;                             // 0 tanh, 1 relu, 2 decay, 3 clamp
};

__global__ void conv_up_kernel(ConvArgs A) {
  __shared__ _Float16 bufA[C_SZ * NFRAMES], bufB[C_SZ * NFRAMES];
  __shared__ float xrow[C_SZ];
  const int b = blockIdx.x, tid = threadIdx.x;
  const int lane = tid & 31, wv = tid >> 5;
  const int half = lane >> 4, mr = lane & 15, nn = lane & 15;

  if (tid < C_SZ) xrow[tid] = A.x[b * C_SZ + tid];
  __syncthreads();

  // Linear 128 -> 1024, reshape (128 ch, 8)
  for (int o = tid; o < C_SZ * 8; o += 256) {
    float acc = A.lin_b[o];
    for (int k = 0; k < C_SZ; ++k) acc += xrow[k] * A.lin_w[k * (C_SZ * 8) + o];
    bufA[(o >> 3) * NFRAMES + (o & 7)] = (_Float16)acc;
  }
  __syncthreads();

  // 4 stages: L 8->16->32->64->128; wave wv owns cout tile [wv*16, wv*16+16)
  for (int s = 0; s < 4; ++s) {
    const int Lu = 16 << s;
    const int ntiles = Lu >> 4;
    const _Float16* pin = (s & 1) ? bufB : bufA;
    _Float16* pout = (s & 1) ? bufA : bufB;
    const float* cw = A.cw[s]; const float* cb = A.cb[s];
    const int co = wv * 16 + mr;
    v8f acc[8];
#pragma unroll
    for (int nt = 0; nt < 8; ++nt) acc[nt] = (v8f){};
    for (int d = 0; d < 3; ++d) {
      for (int ks = 0; ks < 4; ++ks) {
        v16h a;
#pragma unroll
        for (int i = 0; i < 16; ++i)
          a[i] = (_Float16)cw[(co * C_SZ + ks * 32 + a16_k(i, half)) * 3 + d];
        for (int nt = 0; nt < ntiles; ++nt) {
          v16h bb;
#pragma unroll
          for (int i = 0; i < 16; ++i) {
            const int ci = ks * 32 + b16_k(i, half);
            const int q = nt * 16 + nn + d - 1;          // SAME padding
            bb[i] = (q >= 0 && q < Lu) ? pin[ci * NFRAMES + (q >> 1)]
                                       : (_Float16)0.f;  // up2 via q>>1
          }
          acc[nt] = __builtin_amdgcn_wmma_f32_16x16x32_f16(
              false, a, false, bb, (short)0, acc[nt], false, false);
        }
      }
    }
    for (int nt = 0; nt < ntiles; ++nt) {
#pragma unroll
      for (int r = 0; r < 8; ++r) {
        const int cr = wv * 16 + r + 8 * half;
        pout[cr * NFRAMES + nt * 16 + nn] = (_Float16)leaky(acc[nt][r] + cb[cr]);
      }
    }
    __syncthreads();
  }

  // out conv 128 -> 1 at length 128 (final act buffer is bufA)
  if (tid < NFRAMES) {
    float acc = A.ob[0];
    for (int ci = 0; ci < C_SZ; ++ci)
      for (int d = 0; d < 3; ++d) {
        const int q = tid + d - 1;
        if (q >= 0 && q < NFRAMES)
          acc += A.ow[ci * 3 + d] * (float)bufA[ci * NFRAMES + q];
      }
    float v;
    if      (A.act == 0) v = tanhf(acc);
    else if (A.act == 1) v = fmaxf(acc, 0.f);
    else if (A.act == 2) v = 0.8f + 0.2f * clamp01(acc);
    else                 v = clamp01(acc);
    A.outF[b * NFRAMES + tid] = v;
  }
}

// ===========================================================================
// Kernel 3: per-(b,h) blocked prefix-sum of interpolated tone + sin -> osc f16
// grid = B*17, 256 thr (each thread owns 128 contiguous samples).
// ===========================================================================
__global__ void scan_kernel(const float* __restrict__ f0_val,
                            const float* __restrict__ f0cF,
                            const float* __restrict__ harm,
                            _Float16* __restrict__ osc) {
  __shared__ float fr[NFRAMES];
  __shared__ float tsum[256];
  const int b = blockIdx.x / NCH, h = blockIdx.x % NCH, tid = threadIdx.x;

  if (tid < NFRAMES) {
    const float f0 = f0_val[b];
    const float freq = f0 * (1.f + 0.05f * f0cF[b * NFRAMES + tid]);
    const float v = (h == 0) ? freq : freq * harm[h - 1];
    fr[tid] = (v >= 1.f) ? 0.f : v;        // nyquist zeroing at frame level
  }
  __syncthreads();

  // pass 1: per-thread segment sum of interpolated tone values
  const int t0 = tid * 128;
  float s = 0.f;
  for (int j = 0; j < 128; ++j) {
    float pos = (t0 + j + 0.5f) * (1.f / 256.f) - 0.5f;
    pos = fminf(fmaxf(pos, 0.f), 127.f);
    const int i0 = (int)pos, i1 = min(i0 + 1, 127);
    const float w = pos - (float)i0;
    s += fr[i0] + (fr[i1] - fr[i0]) * w;
  }
  tsum[tid] = s;
  __syncthreads();
  // Hillis-Steele inclusive scan over 256 thread sums
  for (int off = 1; off < 256; off <<= 1) {
    const float add = (tid >= off) ? tsum[tid - off] : 0.f;
    __syncthreads();
    tsum[tid] += add;
    __syncthreads();
  }
  float ph = tsum[tid] - s;                // exclusive prefix

  // pass 2: running phase, sin(pi*x) with mod-2 wrap for f32 accuracy
  _Float16* op = osc + ((size_t)(b * NCH + h)) * NSAMP;
  for (int j = 0; j < 128; ++j) {
    const int t = t0 + j;
    float pos = (t + 0.5f) * (1.f / 256.f) - 0.5f;
    pos = fminf(fmaxf(pos, 0.f), 127.f);
    const int i0 = (int)pos, i1 = min(i0 + 1, 127);
    const float w = pos - (float)i0;
    ph += fr[i0] + (fr[i1] - fr[i0]) * w;
    const float xp = ph - 2.f * floorf(0.5f * ph);
    op[t] = (_Float16)sinf(PI_F * xp);
  }
}

// ===========================================================================
// Kernel 4: fused 64-tap FIR (WMMA) + combine; grid = B*(NSAMP/128), 256 thr.
// Taps (17x64 f16, zero-padded to 32x64) staged via async global->LDS DMA.
// ===========================================================================
__global__ void final_kernel(const _Float16* __restrict__ osc,
                             const float* __restrict__ noise,
                             const float* __restrict__ nlvl,
                             const float* __restrict__ spec,
                             const float* __restrict__ ampF,
                             const float* __restrict__ decF,
                             const float* __restrict__ mixF,
                             float* __restrict__ out) {
  __shared__ _Float16 nz[192];             // noise[t0-63 .. t0+127] * level
  __shared__ _Float16 tapsLds[32 * 64];    // filt rows, rows 17..31 zeroed
  __shared__ float bl[NCH * 128];
  __shared__ float awd0[NFRAMES];
  __shared__ float sfac[NCH];
  const int b = blockIdx.x / (NSAMP / 128);
  const int t0 = (blockIdx.x % (NSAMP / 128)) * 128;
  const int tid = threadIdx.x;
  const int lane = tid & 31, wv = tid >> 5;
  const int half = lane >> 4, mr = lane & 15, nn = lane & 15;

  // async-stage filt taps: 17 rows x 128B, one b128 transfer per lane
  // (GLOBAL_LOAD_ASYNC_TO_LDS_B128, GV mode; tracked by ASYNCcnt)
  const _Float16* taps = osc + (size_t)(b * NCH) * NSAMP;
  if (tid < NCH * 8) {
    const int hh = tid >> 3, seg = tid & 7;
    const _Float16* g = taps + (size_t)hh * NSAMP + seg * 8;
    const unsigned l = (unsigned)(uintptr_t)&tapsLds[hh * 64 + seg * 8];
    asm volatile("global_load_async_to_lds_b128 %0, %1, off"
                 :: "v"(l), "v"(g) : "memory");
  }
  for (int i = tid + NCH * 64; i < 32 * 64; i += 256)   // zero pad rows 17..31
    tapsLds[i] = (_Float16)0.f;

  if (tid < 191) {
    const int g = t0 - 63 + tid;
    nz[tid] = (_Float16)((g >= 0) ? noise[b * NSAMP + g] * nlvl[0] : 0.f);
  }
  if (tid < NFRAMES) {                     // awd frame table (channel-0 form)
    const float a  = ampF[b * NFRAMES + tid];
    const float ap = (tid > 0) ? ampF[b * NFRAMES + tid - 1] : 0.f;
    awd0[tid] = a + ap * decF[b * NFRAMES + tid];
  }
  if (tid < NCH) sfac[tid] = (tid == 0) ? 1.f : spec[b * 16 + tid - 1];

  asm volatile("s_wait_asynccnt 0x0" ::: "memory");
  __syncthreads();

  // hint next tile's osc reads into cache (global_prefetch_b8)
  if (t0 + 128 < NSAMP)
    __builtin_prefetch(osc + ((size_t)(b * NCH) * NSAMP) + t0 + 128 + lane * 64, 0, 1);

  // FIR: D[h, t] = sum_{k<64} filt[h,k] * noise[t-k]; wave wv owns n-tile wv
  for (int mt = 0; mt < 2; ++mt) {
    v8f acc = {};
    for (int ks = 0; ks < 2; ++ks) {
      v16h a, bb;
#pragma unroll
      for (int i = 0; i < 16; ++i) {
        a[i] = tapsLds[(mt * 16 + mr) * 64 + ks * 32 + a16_k(i, half)];
        bb[i] = nz[wv * 16 + nn - (ks * 32 + b16_k(i, half)) + 63]; // Toeplitz
      }
      acc = __builtin_amdgcn_wmma_f32_16x16x32_f16(false, a, false, bb,
                                                   (short)0, acc, false, false);
    }
#pragma unroll
    for (int r = 0; r < 8; ++r) {
      const int hh = mt * 16 + r + 8 * half;
      if (hh < NCH) bl[hh * 128 + wv * 16 + nn] = acc[r];
    }
  }
  __syncthreads();

  // combine: full = osc*awd*mix + bl*awd*(1-mix); mean over 17 channels
  if (tid < 128) {
    const int t = t0 + tid;
    float pos = (t + 0.5f) * (1.f / 256.f) - 0.5f;
    pos = fminf(fmaxf(pos, 0.f), 127.f);
    const int i0 = (int)pos, i1 = min(i0 + 1, 127);
    const float w = pos - (float)i0;
    const float mx = mixF[b * NFRAMES + i0] +
                     (mixF[b * NFRAMES + i1] - mixF[b * NFRAMES + i0]) * w;
    const float aw = awd0[i0] + (awd0[i1] - awd0[i0]) * w;
    float acc = 0.f;
    for (int h = 0; h < NCH; ++h) {
      const float awd = sfac[h] * aw;
      const float ov = (float)osc[((size_t)(b * NCH + h)) * NSAMP + t];
      acc += ov * awd * mx + bl[h * 128 + tid] * awd * (1.f - mx);
    }
    out[b * NSAMP + t] = acc * (1.f / (float)NCH);
  }
}

// ===========================================================================
extern "C" void kernel_launch(void* const* d_in, const int* in_sizes, int n_in,
                              void* d_out, int out_size, void* d_ws, size_t ws_size,
                              hipStream_t stream) {
  (void)in_sizes; (void)n_in; (void)out_size; (void)ws_size;
  char* ws = (char*)d_ws;
  float* f0_val = (float*)(ws + WS_F0VAL);
  float* specF  = (float*)(ws + WS_SPEC);
  float* f0cF   = (float*)(ws + WS_F0CF);
  float* ampF   = (float*)(ws + WS_AMPF);
  float* decF   = (float*)(ws + WS_DECF);
  float* mixF   = (float*)(ws + WS_MIXF);
  _Float16* osc = (_Float16*)(ws + WS_OSC);

  // K1: MLP heads
  MlpArgs M;
  M.x = (const float*)d_in[IN_X];
  for (int i = 0; i < 4; ++i) {
    M.fw[i] = (const float*)d_in[IN_F0_W(i)];
    M.fb[i] = (const float*)d_in[IN_F0_B(i)];
    M.sw[i] = (const float*)d_in[IN_SS_W(i)];
    M.sb[i] = (const float*)d_in[IN_SS_B(i)];
  }
  M.gumbel = (const float*)d_in[IN_GUMBEL];
  M.cfreq  = (const float*)d_in[IN_CFREQ];
  M.f0_val = f0_val; M.spec = specF;
  hipLaunchKernelGGL(mlp_kernel, dim3(1), dim3(256), 0, stream, M);

  // K2: 4 conv_upsample heads
  float* frameOut[4] = {f0cF, ampF, decF, mixF};
  for (int h = 0; h < 4; ++h) {
    ConvArgs Cg;
    const int base = IN_CU_BASE(h);
    Cg.x     = (const float*)d_in[IN_X];
    Cg.lin_w = (const float*)d_in[base + 0];
    Cg.lin_b = (const float*)d_in[base + 1];
    for (int i = 0; i < 4; ++i) {
      Cg.cw[i] = (const float*)d_in[base + 2 + i];
      Cg.cb[i] = (const float*)d_in[base + 6 + i];
    }
    Cg.ow = (const float*)d_in[base + 10];
    Cg.ob = (const float*)d_in[base + 11];
    Cg.outF = frameOut[h];
    Cg.act  = h;                       // 0 tanh, 1 relu, 2 decay, 3 clamp
    hipLaunchKernelGGL(conv_up_kernel, dim3(B_SZ), dim3(256), 0, stream, Cg);
  }

  // K3: oscillator bank (cumsum + sin) -> osc f16
  hipLaunchKernelGGL(scan_kernel, dim3(B_SZ * NCH), dim3(256), 0, stream,
                     f0_val, f0cF, (const float*)d_in[IN_HARM], osc);

  // K4: fused FIR + combine -> d_out
  hipLaunchKernelGGL(final_kernel, dim3(B_SZ * (NSAMP / 128)), dim3(256), 0, stream,
                     (const _Float16*)osc, (const float*)d_in[IN_NOISE],
                     (const float*)d_in[IN_NOISE_LVL], specF,
                     ampF, decF, mixF, (float*)d_out);
}
